// CustomBertModel_30408368455757
// MI455X (gfx1250) — matrix operand order repaired and checked by reading, not compiled, not run on hardware
//
#include <hip/hip_runtime.h>
#include <hip/hip_bf16.h>

// ---------------------------------------------------------------------------
// BERT forward on gfx1250 (MI455X).
// bf16 WMMA GEMMs (f32 accumulate), double-buffered async global->LDS staging
// (ASYNCcnt pipeline), weights pre-transposed to [N][K] bf16.
// ---------------------------------------------------------------------------

#define DEV __device__ __forceinline__

constexpr int BATCH = 4;
constexpr int SEQ   = 1024;
constexpr int DIM   = 768;
constexpr int NH    = 12;
constexpr int HD    = 64;
constexpr int FF    = 3072;
constexpr int NL    = 4;
constexpr int TOK   = BATCH * SEQ;   // 4096 token rows

typedef __attribute__((ext_vector_type(16))) __bf16 v16bf;
typedef __attribute__((ext_vector_type(8)))  __bf16 v8bf;
typedef __attribute__((ext_vector_type(8)))  float  v8f;

DEV __bf16 f2bf(float f) {
    unsigned u = __builtin_bit_cast(unsigned, f);
    unsigned r = 0x7FFFu + ((u >> 16) & 1u);           // round to nearest even
    unsigned short s = (unsigned short)((u + r) >> 16);
    return __builtin_bit_cast(__bf16, s);
}

// async 16B global -> LDS copy (per active lane). VDST = LDS byte address,
// VADDR = 64-bit global address. Tracked by ASYNCcnt (in-order completion).
DEV void async_g2l_b128(unsigned lds_addr, unsigned long long gaddr) {
    asm volatile("global_load_async_to_lds_b128 %0, %1, off"
                 :: "v"(lds_addr), "v"(gaddr) : "memory");
}
DEV void wait_async6() { asm volatile("s_wait_asynccnt 0x6" ::: "memory"); }
DEV void wait_async0() { asm volatile("s_wait_asynccnt 0x0" ::: "memory"); }
DEV unsigned lds_addr_of(const void* p) {
    // generic pointers to LDS carry the LDS byte offset in their low 32 bits
    return (unsigned)(size_t)p;
}

// ---------------------------------------------------------------------------
// fp32 [K x N] -> bf16 [N x K] transposed conversion (LDS-tiled, 32x32)
// grid: (N/32, K/32, L)
// ---------------------------------------------------------------------------
__global__ void k_f32_to_bf16_T(const float* __restrict__ src,
                                __bf16* __restrict__ dst, int K, int N) {
    __shared__ float tile[32][33];
    size_t base = (size_t)blockIdx.z * K * N;
    int k0 = blockIdx.y * 32, n0 = blockIdx.x * 32;
    int tx = threadIdx.x & 31, ty = threadIdx.x >> 5;   // ty: 0..7
#pragma unroll
    for (int i = 0; i < 4; i++) {
        int k = k0 + ty + i * 8;
        tile[ty + i * 8][tx] = src[base + (size_t)k * N + (n0 + tx)];
    }
    __syncthreads();
#pragma unroll
    for (int i = 0; i < 4; i++) {
        int n = n0 + ty + i * 8;
        dst[base + (size_t)n * K + (k0 + tx)] = f2bf(tile[tx][ty + i * 8]);
    }
}

// ---------------------------------------------------------------------------
// block reduction helper (256 threads)
// ---------------------------------------------------------------------------
DEV float block_sum(float v, float* red) {
    red[threadIdx.x] = v;
    __syncthreads();
    for (int s = 128; s > 0; s >>= 1) {
        if (threadIdx.x < (unsigned)s) red[threadIdx.x] += red[threadIdx.x + s];
        __syncthreads();
    }
    float r = red[0];
    __syncthreads();
    return r;
}

// ---------------------------------------------------------------------------
// embedding gather + LayerNorm (one block per token row)
// ---------------------------------------------------------------------------
__global__ void k_embed_ln(const int* __restrict__ x,
                           const float* __restrict__ we,
                           const float* __restrict__ pe,
                           const float* __restrict__ te,
                           const float* __restrict__ g,
                           const float* __restrict__ b,
                           float* __restrict__ h32,
                           __bf16* __restrict__ h16) {
    __shared__ float red[256];
    int row = blockIdx.x;
    int s   = row % SEQ;
    int tok = x[row];
    float v[3];
    float sum = 0.f;
#pragma unroll
    for (int i = 0; i < 3; i++) {
        int c = threadIdx.x + i * 256;
        float val = we[(size_t)tok * DIM + c] + pe[(size_t)s * DIM + c] + te[c];
        v[i] = val;
        sum += val;
    }
    sum = block_sum(sum, red);
    float mu = sum / (float)DIM;
    float var = 0.f;
#pragma unroll
    for (int i = 0; i < 3; i++) { float d = v[i] - mu; var += d * d; }
    var = block_sum(var, red) / (float)DIM;
    float rs = rsqrtf(var + 1e-12f);
#pragma unroll
    for (int i = 0; i < 3; i++) {
        int c = threadIdx.x + i * 256;
        float o = (v[i] - mu) * rs * g[c] + b[c];
        h32[(size_t)row * DIM + c] = o;
        h16[(size_t)row * DIM + c] = f2bf(o);
    }
}

// ---------------------------------------------------------------------------
// LayerNorm over rows of DIM
// ---------------------------------------------------------------------------
__global__ void k_ln(const float* __restrict__ in,
                     const float* __restrict__ g,
                     const float* __restrict__ b,
                     float* __restrict__ h32,
                     __bf16* __restrict__ h16) {
    __shared__ float red[256];
    int row = blockIdx.x;
    const float* p = in + (size_t)row * DIM;
    float v[3];
    float sum = 0.f;
#pragma unroll
    for (int i = 0; i < 3; i++) { v[i] = p[threadIdx.x + i * 256]; sum += v[i]; }
    sum = block_sum(sum, red);
    float mu = sum / (float)DIM;
    float var = 0.f;
#pragma unroll
    for (int i = 0; i < 3; i++) { float d = v[i] - mu; var += d * d; }
    var = block_sum(var, red) / (float)DIM;
    float rs = rsqrtf(var + 1e-12f);
#pragma unroll
    for (int i = 0; i < 3; i++) {
        int c = threadIdx.x + i * 256;
        float o = (v[i] - mu) * rs * g[c] + b[c];
        h32[(size_t)row * DIM + c] = o;
        h16[(size_t)row * DIM + c] = f2bf(o);
    }
}

// ---------------------------------------------------------------------------
// row softmax over SEQ=1024, f32 in -> bf16 probs out
// ---------------------------------------------------------------------------
__global__ void k_softmax(const float* __restrict__ sc,
                          __bf16* __restrict__ pr) {
    __shared__ float red[256];
    size_t base = ((size_t)blockIdx.y * SEQ + blockIdx.x) * SEQ;
    const float* p = sc + base;
    __bf16* q = pr + base;
    float v[4];
    float mx = -1e30f;
#pragma unroll
    for (int i = 0; i < 4; i++) {
        v[i] = p[threadIdx.x + i * 256];
        mx = fmaxf(mx, v[i]);
    }
    red[threadIdx.x] = mx;
    __syncthreads();
    for (int s = 128; s > 0; s >>= 1) {
        if (threadIdx.x < (unsigned)s)
            red[threadIdx.x] = fmaxf(red[threadIdx.x], red[threadIdx.x + s]);
        __syncthreads();
    }
    mx = red[0];
    __syncthreads();
    float sum = 0.f;
#pragma unroll
    for (int i = 0; i < 4; i++) { v[i] = __expf(v[i] - mx); sum += v[i]; }
    sum = block_sum(sum, red);
    float inv = 1.f / sum;
#pragma unroll
    for (int i = 0; i < 4; i++)
        q[threadIdx.x + i * 256] = f2bf(v[i] * inv);
}

// ---------------------------------------------------------------------------
// bf16 WMMA GEMM:  out = ACT(scale * (A @ B') + bias + resid)
//   A:[M][K] (lda), B:[N][K] (ldb, pre-transposed). Block tile 128x64, BK=64.
//   8 waves, each owns a 32x32 sub-block = 2x2 WMMA tiles; 2 k-substeps per
//   stage -> 8 v_wmma per stage. Double-buffered async global->LDS staging:
//   stage t+1 is issued before waiting on stage t (s_wait_asynccnt 6, in-order
//   completion), overlapping DMA with WMMA.
//   ACT: 0=none 1=gelu(erf) 2=tanh
//   OUTMODE: 0=normal [z][M][N]   1=split-heads  ([TOK,DIM]->[B,H,S,HD])
//            2=merge-heads (z=(b,h): [S,HD]->[B,S,DIM])
//            3=split-heads-transposed ([TOK,DIM]->[B,H,HD,S], for V)
// ---------------------------------------------------------------------------
template <int ACT, int OUTMODE>
__global__ void k_gemm(int M, int N, int K,
                       const __bf16* __restrict__ A, int lda, long long sA,
                       const __bf16* __restrict__ B, int ldb, long long sB,
                       const float* __restrict__ bias,
                       const float* __restrict__ resid,
                       float scale,
                       float* __restrict__ outF,
                       __bf16* __restrict__ outH,
                       int ldc, long long sC) {
    constexpr int BM = 128, BN = 64, BK = 64;
    constexpr int BUFE = (BM + BN) * BK;                  // 12288 elems / buffer
    __shared__ alignas(16) __bf16 smem[2 * BUFE];         // 48 KB

    const int z = blockIdx.z;
    const __bf16* Az = A + (size_t)z * sA;
    const __bf16* Bz = B + (size_t)z * sB;
    const int bm = blockIdx.y * BM;
    const int bn = blockIdx.x * BN;
    const int tid  = threadIdx.x;
    const int wave = tid >> 5;
    const int lane = tid & 31;
    const int wm = wave & 3;          // 4 wave-rows of 32
    const int wn = wave >> 2;         // 2 wave-cols of 32
    const int half = lane >> 4;
    const int l16  = lane & 15;

    // staging chunk map: 16B (8 bf16) per chunk, 8 chunks per 64-wide row.
    // A: 1024 chunks (4/thread), B: 512 chunks (2/thread). Row indices are
    // clamped (not predicated) so every wave issues exactly 6 asyncs/stage.
    const int ac  = tid >> 3;             // 0..31
    const int akk = (tid & 7) * 8;
    int ar[4], gra[4];
#pragma unroll
    for (int i = 0; i < 4; i++) {
        ar[i]  = ac + i * 32;             // 0..127
        int gr = bm + ar[i];
        gra[i] = gr < M ? gr : (M - 1);   // clamp pooler tail, keep alignment
    }
    const int bnn0 = ac;                  // 0..31
    const int bnn1 = ac + 32;             // 32..63

    const unsigned lbase = lds_addr_of(smem);
    constexpr unsigned BUFB = BUFE * 2;   // bytes per buffer
    unsigned laA[4], laB[2];
#pragma unroll
    for (int i = 0; i < 4; i++) laA[i] = lbase + (ar[i] * BK + akk) * 2;
    laB[0] = lbase + (BM * BK + bnn0 * BK + akk) * 2;
    laB[1] = lbase + (BM * BK + bnn1 * BK + akk) * 2;

    unsigned long long gA[4], gB[2];
#pragma unroll
    for (int i = 0; i < 4; i++)
        gA[i] = (unsigned long long)(size_t)(Az + (size_t)gra[i] * lda + akk);
    gB[0] = (unsigned long long)(size_t)(Bz + (size_t)(bn + bnn0) * ldb + akk);
    gB[1] = (unsigned long long)(size_t)(Bz + (size_t)(bn + bnn1) * ldb + akk);

    v8f acc[2][2];
#pragma unroll
    for (int i = 0; i < 2; i++)
#pragma unroll
        for (int j = 0; j < 2; j++)
            acc[i][j] = (v8f){0.f, 0.f, 0.f, 0.f, 0.f, 0.f, 0.f, 0.f};

    auto issue_stage = [&](int buf) {
        const unsigned o = (unsigned)buf * BUFB;
#pragma unroll
        for (int i = 0; i < 4; i++) { async_g2l_b128(laA[i] + o, gA[i]); gA[i] += BK * 2; }
        async_g2l_b128(laB[0] + o, gB[0]); gB[0] += BK * 2;
        async_g2l_b128(laB[1] + o, gB[1]); gB[1] += BK * 2;
    };

    const int T = K / BK;
    issue_stage(0);                                  // prologue
    for (int t = 0; t < T; t++) {
        const int buf = t & 1;
        if (t + 1 < T) {
            issue_stage(buf ^ 1);                    // overlap next stage DMA
            wait_async6();                           // stage t's 6 ops done
        } else {
            wait_async0();
        }
        __syncthreads();                             // all waves' tiles visible

        const __bf16* Asb = smem + buf * BUFE;
        const __bf16* Bsb = Asb + BM * BK;
#pragma unroll
        for (int kk = 0; kk < BK; kk += 32) {
            // CDNA5 bf16 fragment layouts: A lane m=l16, elems0..7=K half*8..,
            // elems8..15=K 16+half*8.. ; B lane n=l16, same K order as A
            union { v16bf v; v8bf h[2]; } a0, a1;
            const int m0 = wm * 32 + l16;
            a0.h[0] = *(const v8bf*)&Asb[m0 * BK + kk + half * 8];
            a0.h[1] = *(const v8bf*)&Asb[m0 * BK + kk + half * 8 + 16];
            a1.h[0] = *(const v8bf*)&Asb[(m0 + 16) * BK + kk + half * 8];
            a1.h[1] = *(const v8bf*)&Asb[(m0 + 16) * BK + kk + half * 8 + 16];
            const int n0 = wn * 32 + l16;
            union { v16bf v; v8bf h[2]; } bu0, bu1;
            bu0.h[0] = *(const v8bf*)&Bsb[n0 * BK + kk + half * 8];
            bu0.h[1] = *(const v8bf*)&Bsb[n0 * BK + kk + half * 8 + 16];
            bu1.h[0] = *(const v8bf*)&Bsb[(n0 + 16) * BK + kk + half * 8];
            bu1.h[1] = *(const v8bf*)&Bsb[(n0 + 16) * BK + kk + half * 8 + 16];
            v16bf b0 = bu0.v;
            v16bf b1 = bu1.v;

            acc[0][0] = __builtin_amdgcn_wmma_f32_16x16x32_bf16(false, a0.v, false, b0, (short)0, acc[0][0], false, false);
            acc[0][1] = __builtin_amdgcn_wmma_f32_16x16x32_bf16(false, a0.v, false, b1, (short)0, acc[0][1], false, false);
            acc[1][0] = __builtin_amdgcn_wmma_f32_16x16x32_bf16(false, a1.v, false, b0, (short)0, acc[1][0], false, false);
            acc[1][1] = __builtin_amdgcn_wmma_f32_16x16x32_bf16(false, a1.v, false, b1, (short)0, acc[1][1], false, false);
        }
        __syncthreads();                 // reads done before t+2 overwrites buf
    }

    // epilogue: D elem v -> row tileM + v + half*8, col tileN + l16
#pragma unroll
    for (int ti = 0; ti < 2; ti++) {
#pragma unroll
        for (int tj = 0; tj < 2; tj++) {
            const int col = bn + wn * 32 + tj * 16 + l16;
#pragma unroll
            for (int v = 0; v < 8; v++) {
                int row = bm + wm * 32 + ti * 16 + v + half * 8;
                if (row >= M || col >= N) continue;
                float val = acc[ti][tj][v] * scale;
                if (bias) val += bias[col];
                size_t dst;
                if (OUTMODE == 0) {
                    dst = (size_t)z * sC + (size_t)row * ldc + col;
                } else if (OUTMODE == 1) {
                    int b = row / SEQ, s = row % SEQ;
                    int h = col / HD,  d = col % HD;
                    dst = (((size_t)(b * NH + h)) * SEQ + s) * HD + d;
                } else if (OUTMODE == 2) {
                    int b = z / NH, h = z % NH;
                    dst = ((size_t)(b * SEQ + row)) * DIM + h * HD + col;
                } else {
                    int b = row / SEQ, s = row % SEQ;
                    int h = col / HD,  d = col % HD;
                    dst = (((size_t)(b * NH + h)) * HD + d) * SEQ + s;
                }
                if (resid) val += resid[dst];
                if (ACT == 1) val = 0.5f * val * (1.0f + erff(val * 0.70710678118654752f));
                else if (ACT == 2) val = tanhf(val);
                if (outF) outF[dst] = val;
                if (outH) outH[dst] = f2bf(val);
            }
        }
    }
}

// ---------------------------------------------------------------------------
// host-side dispatch
// ---------------------------------------------------------------------------
static void launch_gemm(hipStream_t st, int act, int outmode,
                        int M, int N, int K,
                        const __bf16* A, int lda, long long sA,
                        const __bf16* B, int ldb, long long sB,
                        const float* bias, const float* resid, float scale,
                        float* outF, __bf16* outH, int ldc, long long sC, int Z) {
    dim3 g((N + 63) / 64, (M + 127) / 128, Z), blk(256);
    if (act == 0 && outmode == 0)
        k_gemm<0, 0><<<g, blk, 0, st>>>(M, N, K, A, lda, sA, B, ldb, sB, bias, resid, scale, outF, outH, ldc, sC);
    else if (act == 0 && outmode == 1)
        k_gemm<0, 1><<<g, blk, 0, st>>>(M, N, K, A, lda, sA, B, ldb, sB, bias, resid, scale, outF, outH, ldc, sC);
    else if (act == 0 && outmode == 2)
        k_gemm<0, 2><<<g, blk, 0, st>>>(M, N, K, A, lda, sA, B, ldb, sB, bias, resid, scale, outF, outH, ldc, sC);
    else if (act == 0 && outmode == 3)
        k_gemm<0, 3><<<g, blk, 0, st>>>(M, N, K, A, lda, sA, B, ldb, sB, bias, resid, scale, outF, outH, ldc, sC);
    else if (act == 1)
        k_gemm<1, 0><<<g, blk, 0, st>>>(M, N, K, A, lda, sA, B, ldb, sB, bias, resid, scale, outF, outH, ldc, sC);
    else
        k_gemm<2, 0><<<g, blk, 0, st>>>(M, N, K, A, lda, sA, B, ldb, sB, bias, resid, scale, outF, outH, ldc, sC);
}

static void convertT(hipStream_t st, const float* src, __bf16* dst, int K, int N, int L) {
    k_f32_to_bf16_T<<<dim3(N / 32, K / 32, L), 256, 0, st>>>(src, dst, K, N);
}

extern "C" void kernel_launch(void* const* d_in, const int* in_sizes, int n_in,
                              void* d_out, int out_size, void* d_ws, size_t ws_size,
                              hipStream_t stream) {
    (void)in_sizes; (void)n_in; (void)out_size; (void)ws_size;
    const int*   x    = (const int*)  d_in[0];
    const float* we   = (const float*)d_in[1];
    const float* pe   = (const float*)d_in[2];
    const float* te   = (const float*)d_in[3];
    const float* eg   = (const float*)d_in[4];
    const float* eb   = (const float*)d_in[5];
    const float* Wq   = (const float*)d_in[6];
    const float* bq   = (const float*)d_in[7];
    const float* Wk   = (const float*)d_in[8];
    const float* bk   = (const float*)d_in[9];
    const float* Wv   = (const float*)d_in[10];
    const float* bv   = (const float*)d_in[11];
    const float* Wo   = (const float*)d_in[12];
    const float* bo   = (const float*)d_in[13];
    const float* l1g  = (const float*)d_in[14];
    const float* l1b  = (const float*)d_in[15];
    const float* W1   = (const float*)d_in[16];
    const float* b1   = (const float*)d_in[17];
    const float* W2   = (const float*)d_in[18];
    const float* b2   = (const float*)d_in[19];
    const float* l2g  = (const float*)d_in[20];
    const float* l2b  = (const float*)d_in[21];
    const float* Wp   = (const float*)d_in[22];
    const float* bp   = (const float*)d_in[23];

    // ---- workspace layout ----
    char* ws = (char*)d_ws;
    size_t off = 0;
    auto take = [&](size_t bytes) -> char* {
        char* p = ws + off;
        off += (bytes + 255) & ~(size_t)255;
        return p;
    };
    // all weight copies are TRANSPOSED: [N][K] bf16
    __bf16* wq16 = (__bf16*)take((size_t)NL * DIM * DIM * 2);
    __bf16* wk16 = (__bf16*)take((size_t)NL * DIM * DIM * 2);
    __bf16* wv16 = (__bf16*)take((size_t)NL * DIM * DIM * 2);
    __bf16* wo16 = (__bf16*)take((size_t)NL * DIM * DIM * 2);
    __bf16* w116 = (__bf16*)take((size_t)NL * DIM * FF * 2);   // [FF][DIM] per layer
    __bf16* w216 = (__bf16*)take((size_t)NL * FF * DIM * 2);   // [DIM][FF] per layer
    __bf16* wp16 = (__bf16*)take((size_t)DIM * DIM * 2);
    float*  h32  = (float*) take((size_t)TOK * DIM * 4);
    __bf16* h16  = (__bf16*)take((size_t)TOK * DIM * 2);
    __bf16* q16  = (__bf16*)take((size_t)TOK * DIM * 2);       // [B,H,S,HD]
    __bf16* k16  = (__bf16*)take((size_t)TOK * DIM * 2);       // [B,H,S,HD]
    __bf16* v16  = (__bf16*)take((size_t)TOK * DIM * 2);       // [B,H,HD,S] (transposed)
    float*  sc32 = (float*) take((size_t)BATCH * NH * SEQ * SEQ * 4);
    __bf16* pr16 = (__bf16*)take((size_t)BATCH * NH * SEQ * SEQ * 2);
    __bf16* at16 = (__bf16*)take((size_t)TOK * DIM * 2);
    float*  t32  = (float*) take((size_t)TOK * DIM * 4);
    __bf16* in16 = (__bf16*)take((size_t)TOK * FF * 2);

    // ---- weights fp32 -> bf16 transposed (every call: deterministic) ----
    convertT(stream, Wq, wq16, DIM, DIM, NL);
    convertT(stream, Wk, wk16, DIM, DIM, NL);
    convertT(stream, Wv, wv16, DIM, DIM, NL);
    convertT(stream, Wo, wo16, DIM, DIM, NL);
    convertT(stream, W1, w116, DIM, FF, NL);
    convertT(stream, W2, w216, FF, DIM, NL);
    convertT(stream, Wp, wp16, DIM, DIM, 1);

    // ---- embeddings + LN ----
    k_embed_ln<<<TOK, 256, 0, stream>>>(x, we, pe, te, eg, eb, h32, h16);

    const long long sQ = (long long)SEQ * HD;
    const long long sS = (long long)SEQ * SEQ;

    for (int l = 0; l < NL; l++) {
        const __bf16* Wq_l = wq16 + (size_t)l * DIM * DIM;
        const __bf16* Wk_l = wk16 + (size_t)l * DIM * DIM;
        const __bf16* Wv_l = wv16 + (size_t)l * DIM * DIM;
        const __bf16* Wo_l = wo16 + (size_t)l * DIM * DIM;
        const __bf16* W1_l = w116 + (size_t)l * DIM * FF;
        const __bf16* W2_l = w216 + (size_t)l * FF * DIM;

        // Q,K -> [B,H,S,HD]; V -> [B,H,HD,S]
        launch_gemm(stream, 0, 1, TOK, DIM, DIM, h16, DIM, 0, Wq_l, DIM, 0,
                    bq + l * DIM, nullptr, 1.f, nullptr, q16, DIM, 0, 1);
        launch_gemm(stream, 0, 1, TOK, DIM, DIM, h16, DIM, 0, Wk_l, DIM, 0,
                    bk + l * DIM, nullptr, 1.f, nullptr, k16, DIM, 0, 1);
        launch_gemm(stream, 0, 3, TOK, DIM, DIM, h16, DIM, 0, Wv_l, DIM, 0,
                    bv + l * DIM, nullptr, 1.f, nullptr, v16, DIM, 0, 1);

        // scores = (Q @ K^T)/8 : B matrix = K head [n=s][k=d], contiguous in k
        launch_gemm(stream, 0, 0, SEQ, SEQ, HD, q16, HD, sQ, k16, HD, sQ,
                    nullptr, nullptr, 0.125f, sc32, nullptr, SEQ, sS, BATCH * NH);

        // softmax rows -> bf16 probs
        k_softmax<<<dim3(SEQ, BATCH * NH), 256, 0, stream>>>(sc32, pr16);

        // O = P @ V : B matrix = V^T head [n=d][k=s], contiguous in k
        launch_gemm(stream, 0, 2, SEQ, HD, SEQ, pr16, SEQ, sS, v16, SEQ, sQ,
                    nullptr, nullptr, 1.f, nullptr, at16, HD, 0, BATCH * NH);

        // attention out proj + bias + residual, then LN1
        launch_gemm(stream, 0, 0, TOK, DIM, DIM, at16, DIM, 0, Wo_l, DIM, 0,
                    bo + l * DIM, h32, 1.f, t32, nullptr, DIM, 0, 1);
        k_ln<<<TOK, 256, 0, stream>>>(t32, l1g + l * DIM, l1b + l * DIM, h32, h16);

        // FFN: gelu(h@W1+b1) -> gelu(inter@W2+b2) -> LN2 (no residual)
        launch_gemm(stream, 1, 0, TOK, FF, DIM, h16, DIM, 0, W1_l, DIM, 0,
                    b1 + l * FF, nullptr, 1.f, nullptr, in16, FF, 0, 1);
        launch_gemm(stream, 1, 0, TOK, DIM, FF, in16, FF, 0, W2_l, FF, 0,
                    b2 + l * DIM, nullptr, 1.f, t32, nullptr, DIM, 0, 1);
        k_ln<<<TOK, 256, 0, stream>>>(t32, l2g + l * DIM, l2b + l * DIM, h32, h16);
    }

    // pooler: tanh(h[:,0] @ Wp + bp) -> d_out [4,768]; lda=SEQ*DIM picks token 0
    launch_gemm(stream, 2, 0, BATCH, DIM, DIM, h16, SEQ * DIM, 0, wp16, DIM, 0,
                bp, nullptr, 1.f, (float*)d_out, nullptr, DIM, 0, 1);
}
